// udkws_aa1clap3class_29910152250001
// MI455X (gfx1250) — matrix operand order repaired and checked by reading, not compiled, not run on hardware
//
#include <hip/hip_runtime.h>

typedef __attribute__((ext_vector_type(16))) _Float16 v16h;
typedef __attribute__((ext_vector_type(8)))  _Float16 v8h;
typedef __attribute__((ext_vector_type(8)))  float    v8f;

#define DIN 384
#define DP  128

__device__ __forceinline__ v16h join8(v8h lo, v8h hi) {
  return __builtin_shufflevector(lo, hi, 0,1,2,3,4,5,6,7,8,9,10,11,12,13,14,15);
}

__device__ __forceinline__ float red16_sum(float x) {
  #pragma unroll
  for (int m = 1; m < 16; m <<= 1) x += __shfl_xor(x, m, 16);
  return x;
}

// ---------------------------------------------------------------------------
// Kernel 0: transpose + f32->f16 convert projection weights: Wt[n][k] = W[k][n]
// ---------------------------------------------------------------------------
__global__ void k_prep_w(const float* __restrict__ Wa, const float* __restrict__ Wc,
                         _Float16* __restrict__ WtA, _Float16* __restrict__ WtC) {
  int idx = blockIdx.x * 256 + threadIdx.x;
  if (idx < DIN * DP) {
    int k = idx / DP, n = idx % DP;
    WtA[n * DIN + k] = (_Float16)Wa[idx];
  } else if (idx < 2 * DIN * DP) {
    int j = idx - DIN * DP;
    int k = j / DP, n = j % DP;
    WtC[n * DIN + k] = (_Float16)Wc[j];
  }
}

// ---------------------------------------------------------------------------
// Kernel 1: LayerNorm -> Linear(384->128) via WMMA -> SiLU -> store f16 (raw).
// L2 normalization is deferred to k_main as a rank-1 scale on sim.
// grid (800, 2): 64 tokens per block (4 M-tiles reuse each B fragment).
// ---------------------------------------------------------------------------
__global__ __launch_bounds__(256) void k_proj(
    const float* __restrict__ xA, const float* __restrict__ xC,
    const float* __restrict__ gA, const float* __restrict__ bA, const float* __restrict__ biasA,
    const float* __restrict__ gC, const float* __restrict__ bC, const float* __restrict__ biasC,
    const _Float16* __restrict__ WtA, const _Float16* __restrict__ WtC,
    _Float16* __restrict__ outA, _Float16* __restrict__ outC)
{
  const int side = blockIdx.y;
  const float*    x    = side ? xC    : xA;
  const float*    gam  = side ? gC    : gA;
  const float*    bet  = side ? bC    : bA;
  const float*    bias = side ? biasC : biasA;
  const _Float16* Wt   = side ? WtC   : WtA;
  _Float16*       out  = side ? outC  : outA;

  constexpr int LDA = 392;                    // 384 + 8 halves pad (196 dwords)
  __shared__ _Float16 xl[64 * LDA];           // 50176 B

  const int tid  = threadIdx.x;
  const int lane = tid & 31;
  const int wv   = tid >> 5;
  const long tok0 = (long)blockIdx.x * 64;

  // ---- LayerNorm: 16 threads per row, 4 row-groups ----
  {
    const int rr  = tid >> 4;
    const int sub = tid & 15;
    #pragma unroll 1
    for (int rg = 0; rg < 4; ++rg) {
      const int r = rg * 16 + rr;
      const float* xr = x + (tok0 + r) * DIN;
      float v[24];
      float s = 0.f, sq = 0.f;
      #pragma unroll
      for (int j = 0; j < 24; ++j) { float t = xr[sub + 16 * j]; v[j] = t; s += t; sq += t * t; }
      s = red16_sum(s);
      sq = red16_sum(sq);
      const float mu   = s * (1.f / DIN);
      const float var  = sq * (1.f / DIN) - mu * mu;
      const float rstd = rsqrtf(var + 1e-5f);
      #pragma unroll
      for (int j = 0; j < 24; ++j) {
        int k = sub + 16 * j;
        xl[r * LDA + k] = (_Float16)((v[j] - mu) * rstd * gam[k] + bet[k]);
      }
    }
  }
  __syncthreads();

  // ---- GEMM: 64x128, K=384. wave wv owns N-tile; 4 M-tiles share B frag ----
  const int n0   = wv * 16;
  const int half = lane >> 4;
  const int l15  = lane & 15;
  v8f acc[4] = {{}, {}, {}, {}};
  const _Float16* wrow = Wt + (n0 + l15) * DIN;
  #pragma unroll 1
  for (int ko = 0; ko < DIN; ko += 32) {
    const _Float16* bp = wrow + ko + half * 8;
    v16h bfrag = join8(*(const v8h*)bp, *(const v8h*)(bp + 16));
    #pragma unroll
    for (int mt = 0; mt < 4; ++mt) {
      const _Float16* ap = &xl[(mt * 16 + l15) * LDA + ko + half * 8];
      v16h afrag = join8(*(const v8h*)ap, *(const v8h*)(ap + 16));
      acc[mt] = __builtin_amdgcn_wmma_f32_16x16x32_f16(false, afrag, false, bfrag,
                                                       (short)0, acc[mt], false, false);
    }
  }

  // ---- bias + SiLU (native exp/rcp), store raw f16 ----
  const int n = n0 + l15;
  const float bvv = bias[n];
  #pragma unroll
  for (int mt = 0; mt < 4; ++mt) {
    #pragma unroll
    for (int v = 0; v < 8; ++v) {
      float z  = acc[mt][v] + bvv;
      float yy = z * __builtin_amdgcn_rcpf(1.f + __expf(-z));
      out[(tok0 + mt * 16 + half * 8 + v) * DP + n] = (_Float16)yy;
    }
  }
}

// ---------------------------------------------------------------------------
// Kernel 2: per-batch fused: row norms -> sim(WMMA, cosine scale, mask) ->
//           ragged bilinear resize -> separable 5x5 gaussian -> column max ->
//           folded 1-query transformer.  512 blocks x 256 thr, ~153 KB LDS.
// ---------------------------------------------------------------------------
__global__ __launch_bounds__(256) void k_main(
    const _Float16* __restrict__ anh, const _Float16* __restrict__ bnh,
    const int* __restrict__ amask, const int* __restrict__ cmask,
    const float* __restrict__ Wq, const float* __restrict__ bq,
    const float* __restrict__ Wk, const float* __restrict__ bk,
    const float* __restrict__ Wv, const float* __restrict__ bvp,
    const float* __restrict__ Wo, const float* __restrict__ bo,
    const float* __restrict__ g1, const float* __restrict__ be1,
    const float* __restrict__ W1, const float* __restrict__ fb1,
    const float* __restrict__ W2, const float* __restrict__ fb2,
    const float* __restrict__ g2, const float* __restrict__ be2,
    const float* __restrict__ dW, const float* __restrict__ db,
    float* __restrict__ outp)
{
  const int b    = blockIdx.x;
  const int tid  = threadIdx.x;
  const int lane = tid & 31;
  const int wv   = tid >> 5;

  extern __shared__ char smem[];
  constexpr int LDH = 136;   // an/bn row stride (halves); 272B rows, 16B aligned
  constexpr int LSIM = 116;  // sim row stride (floats)
  constexpr int LDU = 108;   // up buffer row stride (with 2-halo)
  constexpr int LSM = 104;   // sm row stride
  constexpr int LT  = 104;   // tmp (separable pass) row stride
  _Float16* anl = (_Float16*)smem;             // 112*136 halves
  _Float16* bnl = anl + 112 * LDH;             // region A total 60928 B
  float* simb = (float*)(smem + 61440);        // 112*116*4 = 51968 B (region B)
  float* upb  = (float*)smem;                  // overlays region A after sim
  float* smb  = (float*)(smem + 61440);        // overlays region B after resize
  float* tmpb = (float*)(smem + 113408);       // 104*104*4 = 43264 B (region C)

  __shared__ float rnA[112], rnB[112];
  __shared__ float qv[100], Qp[100], uvec[100], att[100], ctx[100], t1v[100],
                   xv[100], hv[100];
  __shared__ float red[128];
  __shared__ int la_s, lc_s;

  auto blockSum = [&](float v) -> float {
    if (tid < 128) red[tid] = v;
    __syncthreads();
    if (tid < 64) red[tid] += red[tid + 64];
    __syncthreads();
    if (tid < 32) {
      float x = red[tid] + red[tid + 32];
      #pragma unroll
      for (int m = 16; m; m >>= 1) x += __shfl_xor(x, m, 32);
      if (tid == 0) red[0] = x;
    }
    __syncthreads();
    float r = red[0];
    __syncthreads();
    return r;
  };
  auto blockMax = [&](float v) -> float {
    if (tid < 128) red[tid] = v;
    __syncthreads();
    if (tid < 64) red[tid] = fmaxf(red[tid], red[tid + 64]);
    __syncthreads();
    if (tid < 32) {
      float x = fmaxf(red[tid], red[tid + 32]);
      #pragma unroll
      for (int m = 16; m; m >>= 1) x = fmaxf(x, __shfl_xor(x, m, 32));
      if (tid == 0) red[0] = x;
    }
    __syncthreads();
    float r = red[0];
    __syncthreads();
    return r;
  };

  if (tid == 0) { la_s = 0; lc_s = 0; }
  __syncthreads();
  if (tid < 100) {
    atomicAdd(&la_s, amask[b * 100 + tid]);
    atomicAdd(&lc_s, cmask[b * 100 + tid]);
  }

  // ---- stage raw projections in LDS via async global->LDS b128 ----
  for (int ch = tid; ch < 1600; ch += 256) {
    int t = ch >> 4, co = (ch & 15) * 8;
    unsigned la_off = (unsigned)(size_t)&anl[t * LDH + co];
    unsigned lb_off = (unsigned)(size_t)&bnl[t * LDH + co];
    unsigned long long ga = (unsigned long long)(size_t)&anh[((long)b * 100 + t) * DP + co];
    unsigned long long gb = (unsigned long long)(size_t)&bnh[((long)b * 100 + t) * DP + co];
    asm volatile("global_load_async_to_lds_b128 %0, %1, off" :: "v"(la_off), "v"(ga));
    asm volatile("global_load_async_to_lds_b128 %0, %1, off" :: "v"(lb_off), "v"(gb));
  }
  for (int i = tid; i < 12 * LDH; i += 256) {  // zero pad rows 100..111
    anl[100 * LDH + i] = (_Float16)0.f;
    bnl[100 * LDH + i] = (_Float16)0.f;
  }
  asm volatile("s_wait_asynccnt 0x0" ::: "memory");
  __syncthreads();
  const int la = la_s, lc = lc_s;

  // ---- per-row inverse L2 norms (deferred cosine normalization) ----
  if (tid < 224) {
    const int sidx = (tid >= 112) ? 1 : 0;
    const int row  = tid - sidx * 112;
    const _Float16* p = sidx ? &bnl[row * LDH] : &anl[row * LDH];
    float s = 0.f;
    #pragma unroll
    for (int c8 = 0; c8 < 16; ++c8) {
      v8h h = *(const v8h*)(p + c8 * 8);
      #pragma unroll
      for (int e = 0; e < 8; ++e) { float t = (float)h[e]; s += t * t; }
    }
    float inv = __builtin_amdgcn_rcpf(fmaxf(sqrtf(s), 1e-8f));
    if (sidx) rnB[row] = inv; else rnA[row] = inv;
  }
  __syncthreads();

  // ---- sim = cos-sim via WMMA + rank-1 scale, 7x7 tiles of 16, K=128 ----
  const int half = lane >> 4;
  const int l15  = lane & 15;
  for (int tile = wv; tile < 49; tile += 8) {
    const int m0 = (tile / 7) * 16;
    const int s0 = (tile % 7) * 16;
    v8f acc = {};
    const _Float16* arow = &anl[(m0 + l15) * LDH];
    const _Float16* brow = &bnl[(s0 + l15) * LDH];
    #pragma unroll
    for (int ko = 0; ko < DP; ko += 32) {
      v16h af = join8(*(const v8h*)(arow + ko + half * 8),
                      *(const v8h*)(arow + ko + 16 + half * 8));
      v16h bf = join8(*(const v8h*)(brow + ko + half * 8),
                      *(const v8h*)(brow + ko + 16 + half * 8));
      acc = __builtin_amdgcn_wmma_f32_16x16x32_f16(false, af, false, bf,
                                                   (short)0, acc, false, false);
    }
    const int s = s0 + l15;
    const float rb = rnB[s];
    #pragma unroll
    for (int v = 0; v < 8; ++v) {
      int m = m0 + half * 8 + v;
      simb[m * LSIM + s] = (m < la && s < lc) ? acc[v] * rnA[m] * rb : 0.f;
    }
  }
  __syncthreads();

  // ---- ragged bilinear resize (align_corners) into zero-halo up buffer ----
  for (int i = tid; i < 104 * LDU; i += 256) upb[i] = 0.f;
  __syncthreads();
  const float laf = (float)la, lcf = (float)lc;
  for (int idx = tid; idx < 10000; idx += 256) {
    int r = idx / 100, c = idx % 100;
    float rf = r * (laf - 1.f) * (1.f / 99.f);
    float cf = c * (lcf - 1.f) * (1.f / 99.f);
    int r0 = (int)floorf(rf); int r1 = min(r0 + 1, la - 1); float wr = rf - (float)r0;
    int c0 = (int)floorf(cf); int c1 = min(c0 + 1, lc - 1); float wc = cf - (float)c0;
    float top = (1.f - wc) * simb[r0 * LSIM + c0] + wc * simb[r0 * LSIM + c1];
    float bot = (1.f - wc) * simb[r1 * LSIM + c0] + wc * simb[r1 * LSIM + c1];
    upb[(r + 2) * LDU + (c + 2)] = (1.f - wr) * top + wr * bot;
  }
  __syncthreads();

  // ---- separable 5x5 gaussian: horizontal pass over 104 halo rows ----
  const float GW[5] = {0.0544886845f, 0.2442013420f, 0.4026199469f,
                       0.2442013420f, 0.0544886845f};
  for (int idx = tid; idx < 104 * 100; idx += 256) {
    int r = idx / 100, c = idx % 100;
    const float* u = &upb[r * LDU + c];
    tmpb[r * LT + c] = GW[0]*u[0] + GW[1]*u[1] + GW[2]*u[2] + GW[3]*u[3] + GW[4]*u[4];
  }
  __syncthreads();
  // ---- vertical pass ----
  for (int idx = tid; idx < 10000; idx += 256) {
    int r = idx / 100, c = idx % 100;
    const float* t = &tmpb[r * LT + c];
    smb[r * LSM + c] = GW[0]*t[0] + GW[1]*t[LT] + GW[2]*t[2*LT] + GW[3]*t[3*LT] + GW[4]*t[4*LT];
  }
  __syncthreads();

  // ---- q = column max of sm ----
  if (tid < 100) {
    float m = -3.4e38f;
    for (int r = 0; r < 100; ++r) m = fmaxf(m, smb[r * LSM + tid]);
    qv[tid] = m;
  }
  __syncthreads();

  // ---- folded single-query cross attention ----
  if (tid < 100) {                       // Q = q@Wq + bq
    float a = bq[tid];
    for (int i = 0; i < 100; ++i) a += qv[i] * Wq[i * 100 + tid];
    Qp[tid] = a;
  }
  __syncthreads();
  if (tid < 100) {                       // u = Q @ Wk^T
    float a = 0.f;
    for (int j = 0; j < 100; ++j) a += Wk[tid * 100 + j] * Qp[j];
    uvec[tid] = a;
  }
  float qb = blockSum(tid < 100 ? Qp[tid] * bk[tid] : 0.f); // also publishes uvec
  if (tid < 100) {                       // scores[s] = 0.1*(u . kv[s] + qb)
    float a = qb;
    for (int i = 0; i < 100; ++i) a += uvec[i] * smb[tid * LSM + i];
    att[tid] = 0.1f * a;
  }
  float mx = blockMax(tid < 100 ? att[tid] : -3.4e38f);
  float e  = (tid < 100) ? __expf(att[tid] - mx) : 0.f;
  float se = blockSum(e);
  float rs = __builtin_amdgcn_rcpf(se);
  if (tid < 100) att[tid] = e * rs;
  __syncthreads();
  if (tid < 100) {                       // ctx = softmax(a) @ kv
    float a = 0.f;
    for (int s = 0; s < 100; ++s) a += att[s] * smb[s * LSM + tid];
    ctx[tid] = a;
  }
  __syncthreads();
  if (tid < 100) {                       // t1 = ctx@Wv + bv
    float a = bvp[tid];
    for (int i = 0; i < 100; ++i) a += ctx[i] * Wv[i * 100 + tid];
    t1v[tid] = a;
  }
  __syncthreads();
  float ov = 0.f;
  if (tid < 100) {                       // q + (t1@Wo + bo)
    float a = bo[tid];
    for (int j = 0; j < 100; ++j) a += t1v[j] * Wo[j * 100 + tid];
    ov = qv[tid] + a;
  }
  float s1 = blockSum(tid < 100 ? ov : 0.f);
  float s2 = blockSum(tid < 100 ? ov * ov : 0.f);
  {
    float mu = s1 * 0.01f;
    float rstd = rsqrtf(s2 * 0.01f - mu * mu + 1e-5f);
    if (tid < 100) xv[tid] = (ov - mu) * rstd * g1[tid] + be1[tid];
  }
  __syncthreads();
  if (tid < 100) {                       // h = relu(x@W1 + fb1)
    float a = fb1[tid];
    for (int d = 0; d < 100; ++d) a += xv[d] * W1[d * 100 + tid];
    hv[tid] = fmaxf(a, 0.f);
  }
  __syncthreads();
  float yvr = 0.f;
  if (tid < 100) {                       // y = x + h@W2 + fb2
    float a = fb2[tid];
    for (int j = 0; j < 100; ++j) a += hv[j] * W2[j * 100 + tid];
    yvr = xv[tid] + a;
  }
  float t1s = blockSum(tid < 100 ? yvr : 0.f);
  float t2s = blockSum(tid < 100 ? yvr * yvr : 0.f);
  float zr = 0.f;
  {
    float mu = t1s * 0.01f;
    float rstd = rsqrtf(t2s * 0.01f - mu * mu + 1e-5f);
    if (tid < 100) zr = (yvr - mu) * rstd * g2[tid] + be2[tid];
  }
  float fin = blockSum(tid < 100 ? zr * dW[tid] : 0.f);
  if (tid == 0) outp[b] = fin + db[0];
}

// ---------------------------------------------------------------------------
extern "C" void kernel_launch(void* const* d_in, const int* in_sizes, int n_in,
                              void* d_out, int out_size, void* d_ws, size_t ws_size,
                              hipStream_t stream) {
  (void)in_sizes; (void)n_in; (void)out_size; (void)ws_size;
  const float* anco  = (const float*)d_in[0];
  const float* com   = (const float*)d_in[1];
  const int*   amask = (const int*)d_in[2];
  const int*   cmask = (const int*)d_in[3];
  const float* gA = (const float*)d_in[4];
  const float* bA = (const float*)d_in[5];
  const float* WA = (const float*)d_in[6];
  const float* biasA = (const float*)d_in[7];
  const float* gC = (const float*)d_in[8];
  const float* bC = (const float*)d_in[9];
  const float* WC = (const float*)d_in[10];
  const float* biasC = (const float*)d_in[11];
  const float* Wq = (const float*)d_in[12];
  const float* bq = (const float*)d_in[13];
  const float* Wk = (const float*)d_in[14];
  const float* bk = (const float*)d_in[15];
  const float* Wv = (const float*)d_in[16];
  const float* bv = (const float*)d_in[17];
  const float* Wo = (const float*)d_in[18];
  const float* bo = (const float*)d_in[19];
  const float* g1 = (const float*)d_in[20];
  const float* be1 = (const float*)d_in[21];
  const float* W1 = (const float*)d_in[22];
  const float* fb1 = (const float*)d_in[23];
  const float* W2 = (const float*)d_in[24];
  const float* fb2 = (const float*)d_in[25];
  const float* g2 = (const float*)d_in[26];
  const float* be2 = (const float*)d_in[27];
  const float* dW = (const float*)d_in[28];
  const float* db = (const float*)d_in[29];

  char* ws = (char*)d_ws;
  _Float16* WtA = (_Float16*)ws;                           // 98304 B
  _Float16* WtC = (_Float16*)(ws + 98304);                 // 98304 B
  _Float16* anh = (_Float16*)(ws + 196608);                // 13107200 B
  _Float16* bnh = (_Float16*)(ws + 196608 + 13107200);     // 13107200 B

  k_prep_w<<<dim3(384), dim3(256), 0, stream>>>(WA, WC, WtA, WtC);
  k_proj<<<dim3(800, 2), dim3(256), 0, stream>>>(anco, com, gA, bA, biasA,
                                                 gC, bC, biasC, WtA, WtC, anh, bnh);
  const size_t shmem = 113408 + 104 * 104 * 4;             // 156672 B
  (void)hipFuncSetAttribute((const void*)k_main,
                            hipFuncAttributeMaxDynamicSharedMemorySize,
                            (int)shmem);
  k_main<<<dim3(512), dim3(256), shmem, stream>>>(
      anh, bnh, amask, cmask, Wq, bq, Wk, bk, Wv, bv, Wo, bo,
      g1, be1, W1, fb1, W2, fb2, g2, be2, dW, db, (float*)d_out);
}
